// TGCN_16569983827993
// MI455X (gfx1250) — matrix operand order repaired and checked by reading, not compiled, not run on hardware
//
#include <hip/hip_runtime.h>
#include <hip/hip_bf16.h>
#include <cstdint>
#include <cstddef>

typedef __attribute__((ext_vector_type(16))) _Float16 v16h;
typedef __attribute__((ext_vector_type(4)))  _Float16 v4h;
typedef __attribute__((ext_vector_type(8)))  float    v8f;

#define SLEN 64
#define NN   4096
#define FF   32
#define HH   64
#define OO   2
#define ADJ_SCALE   4096.0f
#define INV_ADJ     (1.0f / 4096.0f)
#define S2_SCALE    256.0f
#define INV_ADJ_S2  (1.0f / (4096.0f * 256.0f))
#define AS 40   // padded LDS stride (halves): 80B -> 20-bank stride, conflict-free b128
#define BS 40
#define NKC (NN / 32)

// Async global->LDS 16B copy (CDNA5 TDM-lite path, ASYNCcnt-tracked).
// LDS dest offset = low 32 bits of the generic shared pointer (ISA 10.2: LDS
// aperture addresses truncate to addr[31:0]).
__device__ __forceinline__ void async_copy16(void* ldsdst, const void* gsrc) {
  uint32_t l = (uint32_t)(uintptr_t)ldsdst;
  asm volatile("global_load_async_to_lds_b128 %0, %1, off"
               :: "v"(l), "v"(gsrc) : "memory");
}
__device__ __forceinline__ void wait_async0() {
  asm volatile("s_wait_asynccnt 0x0" ::: "memory");
}

// ---------------- kernel 0: adjh = f16(adj * 4096) ----------------
__global__ __launch_bounds__(256) void adj_cvt_kernel(const float* __restrict__ adj,
                                                      _Float16* __restrict__ adjh) {
  size_t i = (size_t)blockIdx.x * 256 + threadIdx.x;   // one float4 per thread
  float4 v = reinterpret_cast<const float4*>(adj)[i];
  v4h o;
  o[0] = (_Float16)(v.x * ADJ_SCALE);
  o[1] = (_Float16)(v.y * ADJ_SCALE);
  o[2] = (_Float16)(v.z * ADJ_SCALE);
  o[3] = (_Float16)(v.w * ADJ_SCALE);
  reinterpret_cast<v4h*>(adjh)[i] = o;
}

// ------- kernel 1: supportT[s][h][n] = f16( b1[h] + sum_f x[s][n][f]*W1[f][h] ) -------
__global__ __launch_bounds__(256) void support_kernel(const float* __restrict__ x,
                                                      const float* __restrict__ W1,
                                                      const float* __restrict__ b1,
                                                      _Float16* __restrict__ supportT) {
  __shared__ float W1s[FF * HH];
  __shared__ float b1s[HH];
  int t = threadIdx.x;
  for (int i = t; i < FF * HH; i += 256) W1s[i] = W1[i];
  if (t < HH) b1s[t] = b1[t];
  __syncthreads();

  int g = blockIdx.x * 256 + t;       // (s,n) flat, 262144 total
  int s = g >> 12;
  int n = g & (NN - 1);

  float xr[FF];
  const float4* xp = reinterpret_cast<const float4*>(x + (size_t)g * FF);
#pragma unroll
  for (int q = 0; q < FF / 4; ++q) {
    float4 v = xp[q];
    xr[4 * q + 0] = v.x; xr[4 * q + 1] = v.y;
    xr[4 * q + 2] = v.z; xr[4 * q + 3] = v.w;
  }
  _Float16* outp = supportT + (size_t)s * (HH * NN) + n;
#pragma unroll 4
  for (int h = 0; h < HH; ++h) {
    float acc = b1s[h];
#pragma unroll
    for (int f = 0; f < FF; ++f) acc += xr[f] * W1s[f * HH + h];
    outp[(size_t)h * NN] = (_Float16)acc;   // coalesced across consecutive n
  }
}

// ---- kernel 2: per (i-block, s): C = (4096*adj)[128xK] @ supportT[s] (Kx64) ----
// Double-buffered async-to-LDS staging; 8 waves in 4(m)x2(n) grid, 2x2 tiles each.
// epilogue: h = relu(C/4096); support2[s][i][o] = b2[o] + sum_h h*W2[h][o]
__global__ __launch_bounds__(256) void gemm1_kernel(const _Float16* __restrict__ adjh,
                                                    const _Float16* __restrict__ supportT,
                                                    const float* __restrict__ W2,
                                                    const float* __restrict__ b2,
                                                    float* __restrict__ s2out) {
  __shared__ _Float16 Asub[2][128 * AS];
  __shared__ _Float16 Bsub[2][64 * BS];
  __shared__ float    Ctile[128 * 64];

  int t  = threadIdx.x;
  int i0 = blockIdx.x * 128;
  int s  = blockIdx.y;
  const _Float16* Bglob = supportT + (size_t)s * (HH * NN);

  int lane = t & 31, w = t >> 5, half = lane >> 4, l15 = lane & 15;
  int wm = w & 3, wn = w >> 2;
  int arow = t >> 1;                 // A staging: row, 2 chunks of 8 halves
  int bcol = t >> 2, bkg = t & 3;    // B staging: 1 chunk

  v8f acc[2][2];
#pragma unroll
  for (int mt = 0; mt < 2; ++mt)
#pragma unroll
    for (int nt = 0; nt < 2; ++nt)
#pragma unroll
      for (int r = 0; r < 8; ++r) acc[mt][nt][r] = 0.0f;

  auto stage = [&](int buf, int k0) {
#pragma unroll
    for (int c = 0; c < 2; ++c) {
      int kg = (t & 1) * 2 + c;
      async_copy16(&Asub[buf][arow * AS + kg * 8],
                   adjh + (size_t)(i0 + arow) * NN + k0 + kg * 8);
    }
    async_copy16(&Bsub[buf][bcol * BS + bkg * 8],
                 Bglob + (size_t)bcol * NN + k0 + bkg * 8);
  };

  stage(0, 0);
  wait_async0();
  __syncthreads();

  for (int kc = 0; kc < NKC; ++kc) {
    int cur = kc & 1;
    if (kc + 1 < NKC) stage(cur ^ 1, (kc + 1) * 32);   // prefetch next tile

    const _Float16* Ab = Asub[cur];
    const _Float16* Bb = Bsub[cur];
    v16h a[2], b[2];
#pragma unroll
    for (int mt = 0; mt < 2; ++mt) {
      int row = wm * 32 + mt * 16 + l15;
      reinterpret_cast<uint4*>(&a[mt])[0] =
          *reinterpret_cast<const uint4*>(&Ab[row * AS + half * 8]);
      reinterpret_cast<uint4*>(&a[mt])[1] =
          *reinterpret_cast<const uint4*>(&Ab[row * AS + 16 + half * 8]);
    }
#pragma unroll
    for (int nt = 0; nt < 2; ++nt) {
      int col = wn * 32 + nt * 16 + l15;
      reinterpret_cast<uint4*>(&b[nt])[0] =
          *reinterpret_cast<const uint4*>(&Bb[col * BS + half * 8]);
      reinterpret_cast<uint4*>(&b[nt])[1] =
          *reinterpret_cast<const uint4*>(&Bb[col * BS + 16 + half * 8]);
    }
#pragma unroll
    for (int mt = 0; mt < 2; ++mt)
#pragma unroll
      for (int nt = 0; nt < 2; ++nt)
        acc[mt][nt] = __builtin_amdgcn_wmma_f32_16x16x32_f16(
            false, a[mt], false, b[nt], (short)0, acc[mt][nt], false, false);

    wait_async0();       // our async writes into next buffer landed
    __syncthreads();     // + DScnt wait: everyone done reading current buffer
  }

  // scale + relu into LDS (C layout: vgpr r -> M = r + 8*half, N = l15)
#pragma unroll
  for (int mt = 0; mt < 2; ++mt)
#pragma unroll
    for (int nt = 0; nt < 2; ++nt)
#pragma unroll
      for (int r = 0; r < 8; ++r) {
        float v = acc[mt][nt][r] * INV_ADJ;
        Ctile[(wm * 32 + mt * 16 + r + half * 8) * 64 + wn * 32 + nt * 16 + l15] =
            fmaxf(v, 0.0f);
      }
  __syncthreads();

  // fused h @ W2 + b2 -> support2
  int row = t >> 1, o = t & 1;
  float sacc = b2[o];
#pragma unroll
  for (int n = 0; n < HH; ++n) sacc += Ctile[row * 64 + n] * W2[n * OO + o];
  s2out[(size_t)s * (NN * OO) + (size_t)(i0 + row) * OO + o] = sacc;
}

// ---- kernel 3: g = adj @ support2 over 64 (s,o) columns; epilogue log_softmax ----
__global__ __launch_bounds__(256) void gemm2_kernel(const _Float16* __restrict__ adjh,
                                                    const float* __restrict__ s2,
                                                    float* __restrict__ rnn) {
  __shared__ _Float16 Asub[2][128 * AS];
  __shared__ _Float16 Bsub[2][64 * BS];
  __shared__ float    Ctile[128 * 64];

  int t    = threadIdx.x;
  int i0   = blockIdx.x * 128;
  int col0 = blockIdx.y * 64;        // global col = s*2 + o
  int lane = t & 31, w = t >> 5, half = lane >> 4, l15 = lane & 15;
  int wm = w & 3, wn = w >> 2;
  int arow = t >> 1;

  v8f acc[2][2];
#pragma unroll
  for (int mt = 0; mt < 2; ++mt)
#pragma unroll
    for (int nt = 0; nt < 2; ++nt)
#pragma unroll
      for (int r = 0; r < 8; ++r) acc[mt][nt][r] = 0.0f;

  auto stageA = [&](int buf, int k0) {
#pragma unroll
    for (int c = 0; c < 2; ++c) {
      int kg = (t & 1) * 2 + c;
      async_copy16(&Asub[buf][arow * AS + kg * 8],
                   adjh + (size_t)(i0 + arow) * NN + k0 + kg * 8);
    }
  };
  auto stageB = [&](int buf, int k0) {   // f32 -> f16 (scaled), manual DS stores
    int colL = t & 63;
    int kb   = (t >> 6) * 8;
    int gcol = col0 + colL;
    int ss = gcol >> 1, o = gcol & 1;
#pragma unroll
    for (int u = 0; u < 8; ++u) {
      float v = s2[(size_t)ss * (NN * OO) + (size_t)(k0 + kb + u) * OO + o];
      Bsub[buf][colL * BS + kb + u] = (_Float16)(v * S2_SCALE);
    }
  };

  stageA(0, 0);
  stageB(0, 0);
  wait_async0();
  __syncthreads();

  for (int kc = 0; kc < NKC; ++kc) {
    int cur = kc & 1;
    if (kc + 1 < NKC) { stageA(cur ^ 1, (kc + 1) * 32); stageB(cur ^ 1, (kc + 1) * 32); }

    const _Float16* Ab = Asub[cur];
    const _Float16* Bb = Bsub[cur];
    v16h a[2], b[2];
#pragma unroll
    for (int mt = 0; mt < 2; ++mt) {
      int row = wm * 32 + mt * 16 + l15;
      reinterpret_cast<uint4*>(&a[mt])[0] =
          *reinterpret_cast<const uint4*>(&Ab[row * AS + half * 8]);
      reinterpret_cast<uint4*>(&a[mt])[1] =
          *reinterpret_cast<const uint4*>(&Ab[row * AS + 16 + half * 8]);
    }
#pragma unroll
    for (int nt = 0; nt < 2; ++nt) {
      int col = wn * 32 + nt * 16 + l15;
      reinterpret_cast<uint4*>(&b[nt])[0] =
          *reinterpret_cast<const uint4*>(&Bb[col * BS + half * 8]);
      reinterpret_cast<uint4*>(&b[nt])[1] =
          *reinterpret_cast<const uint4*>(&Bb[col * BS + 16 + half * 8]);
    }
#pragma unroll
    for (int mt = 0; mt < 2; ++mt)
#pragma unroll
      for (int nt = 0; nt < 2; ++nt)
        acc[mt][nt] = __builtin_amdgcn_wmma_f32_16x16x32_f16(
            false, a[mt], false, b[nt], (short)0, acc[mt][nt], false, false);

    wait_async0();
    __syncthreads();
  }

#pragma unroll
  for (int mt = 0; mt < 2; ++mt)
#pragma unroll
    for (int nt = 0; nt < 2; ++nt)
#pragma unroll
      for (int r = 0; r < 8; ++r)
        Ctile[(wm * 32 + mt * 16 + r + half * 8) * 64 + wn * 32 + nt * 16 + l15] =
            acc[mt][nt][r] * INV_ADJ_S2;
  __syncthreads();

  // log_softmax over O=2 (adjacent columns), write rnn_in[s][i*2+o]
  for (int u = 0; u < 16; ++u) {
    int idx = t + 256 * u;
    int row = idx >> 5, sp = idx & 31;
    float c0 = Ctile[row * 64 + 2 * sp];
    float c1 = Ctile[row * 64 + 2 * sp + 1];
    float m  = fmaxf(c0, c1);
    float lse = m + logf(expf(c0 - m) + expf(c1 - m));
    int sg = (col0 >> 1) + sp;
    int gi = i0 + row;
    rnn[(size_t)sg * (NN * OO) + (size_t)gi * OO + 0] = c0 - lse;
    rnn[(size_t)sg * (NN * OO) + (size_t)gi * OO + 1] = c1 - lse;
  }
}

// ---- kernel 4: GI0[s][r] = bih0[r] + rnn[s] . Wih0[r] (K = 8192) ----
__global__ __launch_bounds__(256) void gi0_kernel(const float* __restrict__ rnn,
                                                  const float* __restrict__ Wih0,
                                                  const float* __restrict__ bih0,
                                                  float* __restrict__ GI0) {
  int r = blockIdx.x;     // 0..191
  int s = blockIdx.y;     // 0..63
  int t = threadIdx.x;
  const float* xr = rnn  + (size_t)s * (NN * OO);
  const float* wr = Wih0 + (size_t)r * (NN * OO);
  float acc = 0.0f;
  for (int k = t; k < NN * OO; k += 256) acc += xr[k] * wr[k];
  __shared__ float red[256];
  red[t] = acc;
  __syncthreads();
  for (int off = 128; off > 0; off >>= 1) {
    if (t < off) red[t] += red[t + off];
    __syncthreads();
  }
  if (t == 0) GI0[s * 192 + r] = red[0] + bih0[r];
}

__device__ __forceinline__ float sigm(float x) { return 1.0f / (1.0f + expf(-x)); }

// ---- kernel 5: two-layer GRU recurrence (sequential over S), 1 block x 64 threads ----
__global__ __launch_bounds__(64) void gru_kernel(const float* __restrict__ GI0,
                                                 const float* __restrict__ Whh0,
                                                 const float* __restrict__ bhh0,
                                                 const float* __restrict__ Wih1,
                                                 const float* __restrict__ Whh1,
                                                 const float* __restrict__ bih1,
                                                 const float* __restrict__ bhh1,
                                                 float* __restrict__ out) {
  __shared__ float h[HH];
  __shared__ float ys0[SLEN][HH];
  int t = threadIdx.x;
  h[t] = 0.0f;
  __syncthreads();

  // layer 0 (gi precomputed in GI0)
  for (int s = 0; s < SLEN; ++s) {
    float ghr = bhh0[t], ghz = bhh0[HH + t], ghn = bhh0[2 * HH + t];
#pragma unroll 4
    for (int j = 0; j < HH; ++j) {
      float hj = h[j];
      ghr += Whh0[t * HH + j] * hj;
      ghz += Whh0[(HH + t) * HH + j] * hj;
      ghn += Whh0[(2 * HH + t) * HH + j] * hj;
    }
    float gir = GI0[s * 192 + t], giz = GI0[s * 192 + HH + t], gin = GI0[s * 192 + 2 * HH + t];
    float r = sigm(gir + ghr);
    float z = sigm(giz + ghz);
    float n = tanhf(gin + r * ghn);
    float hn_ = (1.0f - z) * n + z * h[t];
    __syncthreads();
    h[t] = hn_;
    ys0[s][t] = hn_;
    __syncthreads();
  }
  out[SLEN * HH + t] = h[t];        // hn[0]

  __syncthreads();
  h[t] = 0.0f;
  __syncthreads();

  // layer 1
  for (int s = 0; s < SLEN; ++s) {
    float gir = bih1[t], giz = bih1[HH + t], gin = bih1[2 * HH + t];
    float ghr = bhh1[t], ghz = bhh1[HH + t], ghn = bhh1[2 * HH + t];
#pragma unroll 4
    for (int j = 0; j < HH; ++j) {
      float xj = ys0[s][j];
      float hj = h[j];
      gir += Wih1[t * HH + j] * xj;
      giz += Wih1[(HH + t) * HH + j] * xj;
      gin += Wih1[(2 * HH + t) * HH + j] * xj;
      ghr += Whh1[t * HH + j] * hj;
      ghz += Whh1[(HH + t) * HH + j] * hj;
      ghn += Whh1[(2 * HH + t) * HH + j] * hj;
    }
    float r = sigm(gir + ghr);
    float z = sigm(giz + ghz);
    float n = tanhf(gin + r * ghn);
    float hn_ = (1.0f - z) * n + z * h[t];
    __syncthreads();
    h[t] = hn_;
    __syncthreads();
    out[s * HH + t] = hn_;          // ys1
  }
  out[SLEN * HH + HH + t] = h[t];   // hn[1]
}

extern "C" void kernel_launch(void* const* d_in, const int* in_sizes, int n_in,
                              void* d_out, int out_size, void* d_ws, size_t ws_size,
                              hipStream_t stream) {
  const float* x    = (const float*)d_in[0];
  const float* adj  = (const float*)d_in[1];
  const float* W1   = (const float*)d_in[2];
  const float* b1   = (const float*)d_in[3];
  const float* W2   = (const float*)d_in[4];
  const float* b2   = (const float*)d_in[5];
  const float* Wih0 = (const float*)d_in[6];
  const float* Whh0 = (const float*)d_in[7];
  const float* bih0 = (const float*)d_in[8];
  const float* bhh0 = (const float*)d_in[9];
  const float* Wih1 = (const float*)d_in[10];
  const float* Whh1 = (const float*)d_in[11];
  const float* bih1 = (const float*)d_in[12];
  const float* bhh1 = (const float*)d_in[13];
  float* out = (float*)d_out;

  // workspace layout (~68 MB)
  char* ws = (char*)d_ws;
  _Float16* adjh     = (_Float16*)(ws);                                  // 32 MB
  _Float16* supportT = (_Float16*)(ws + (size_t)NN * NN * 2);            // 32 MB
  float*    s2       = (float*)(ws + (size_t)NN * NN * 4);               // 2 MB
  float*    rnn      = (float*)(ws + (size_t)NN * NN * 4 + (size_t)SLEN * NN * OO * 4);
  float*    GI0      = (float*)(ws + (size_t)NN * NN * 4 + (size_t)2 * SLEN * NN * OO * 4);

  adj_cvt_kernel<<<(NN * NN) / (4 * 256), 256, 0, stream>>>(adj, adjh);
  support_kernel<<<(SLEN * NN) / 256, 256, 0, stream>>>(x, W1, b1, supportT);
  gemm1_kernel<<<dim3(NN / 128, SLEN), 256, 0, stream>>>(adjh, supportT, W2, b2, s2);
  gemm2_kernel<<<dim3(NN / 128, (SLEN * OO) / 64), 256, 0, stream>>>(adjh, s2, rnn);
  gi0_kernel<<<dim3(192, SLEN), 256, 0, stream>>>(rnn, Wih0, bih0, GI0);
  gru_kernel<<<1, 64, 0, stream>>>(GI0, Whh0, bhh0, Wih1, Whh1, bih1, bhh1, out);
}
